// ExpMovAvgModel_23845658427637
// MI455X (gfx1250) — compile-verified, matrix operand order checked by reading, not run on hardware
//
#include <hip/hip_runtime.h>

#define B_DIM 16
#define T_DIM 2048
#define K_DIM 64
#define H_DIM 128
#define NTILE (T_DIM / 16)   // 128 16-wide tiles per batch row
#define LOG2E 1.44269504088896340736f

typedef __attribute__((ext_vector_type(16))) __bf16 v16bf;
typedef __attribute__((ext_vector_type(8)))  __bf16 v8bf;
typedef __attribute__((ext_vector_type(8)))  float  v8f;

__device__ __forceinline__ unsigned short f32_to_bf16(float f) {
    unsigned u = __float_as_uint(f);
    u += 0x7FFFu + ((u >> 16) & 1u);       // round-to-nearest-even
    return (unsigned short)(u >> 16);
}

// CDNA5 16-bit A (16x32) fragment layout (CDNA5 ISA §7.12.2):
// lanes 0-15 hold row M=lane, K = {0..7} in v0-3 and {16..23} in v4-7;
// lanes 16-31 hold K = {8..15} and {24..31}. B (32x16) uses the transposed
// identical per-lane pattern (lane -> column n = lane&15).
__device__ __forceinline__ v16bf load_frag(const unsigned short* rowbase, int kh8) {
    const v8bf c0 = *(const v8bf*)(rowbase + kh8);        // k = kh8 .. kh8+7
    const v8bf c1 = *(const v8bf*)(rowbase + kh8 + 16);   // k = kh8+16 .. kh8+23
    v16bf r;
#pragma unroll
    for (int j = 0; j < 8; ++j) { r[j] = c0[j]; r[j + 8] = c1[j]; }
    return r;
}

// ---- Kernel 0: collapse H=128 -> three K=64 projection vectors ----
__global__ void ektc_proj(const float* __restrict__ kcw,  // [64,128]
                          const float* __restrict__ lamw, // [128]
                          const float* __restrict__ gw,   // [128]
                          const float* __restrict__ sw,   // [128]
                          float* __restrict__ vlam, float* __restrict__ vg,
                          float* __restrict__ vs) {
    int k = threadIdx.x;
    if (k >= K_DIM) return;
    float dl = 0.f, dg = 0.f, ds = 0.f;
    for (int h = 0; h < H_DIM; ++h) {
        float w = kcw[k * H_DIM + h];
        dl += w * lamw[h]; dg += w * gw[h]; ds += w * sw[h];
    }
    vlam[k] = dl; vg[k] = dg; vs[k] = ds;
}

// ---- Kernel 1: gather x = A[problem_seq], bf16 convert, per-(b,t) scalars ----
__global__ void ektc_gather(const int* __restrict__ pseq, const float* __restrict__ A,
                            const float* __restrict__ vlam, const float* __restrict__ vg,
                            const float* __restrict__ vs,
                            const float* __restrict__ lamb, const float* __restrict__ gb,
                            const float* __restrict__ sb,
                            unsigned short* __restrict__ xh,   // [B*T,64] bf16 bits
                            float* __restrict__ lam, float* __restrict__ guess,
                            float* __restrict__ slip) {
    int idx = blockIdx.x * blockDim.x + threadIdx.x;
    if (idx >= B_DIM * T_DIM) return;
    const float* ar = A + (size_t)pseq[idx] * K_DIM;
    float dl = 0.f, dg = 0.f, ds = 0.f;
#pragma unroll 8
    for (int k = 0; k < K_DIM; ++k) {
        float v = ar[k];
        xh[(size_t)idx * K_DIM + k] = f32_to_bf16(v);
        dl += v * vlam[k]; dg += v * vg[k]; ds += v * vs[k];
    }
    lam[idx]   = __expf(dl + lamb[0]);
    guess[idx] = 1.0f / (1.0f + __expf(-(dg + gb[0])));
    slip[idx]  = 1.0f / (1.0f + __expf(-(ds + sb[0])));
}

// ---- Kernel 1b: reduction-WMMA A fragments per (b, 16-tile, lane) ----
// A2 row 0 = ones -> sum(w); row 1 = y[s_base+k] -> sum(y*w); rows>=2 zero.
// Element j -> K = 8*(lane>>4) + j  (one 16-row K-slice; blocks concatenate two).
__global__ void ektc_yfrag(const float* __restrict__ y,
                           unsigned short* __restrict__ ayfrag) { // [B*NTILE*32*8]
    int idx = blockIdx.x * blockDim.x + threadIdx.x;   // (b*NTILE + st)*32 + lane
    if (idx >= B_DIM * NTILE * 32) return;
    const int lane = idx & 31;
    const int st   = (idx >> 5) & (NTILE - 1);
    const int b    = idx >> (5 + 7);
    const int l16  = lane & 15;
    const int kh8  = (lane >> 4) * 8;
#pragma unroll
    for (int j = 0; j < 8; ++j) {
        unsigned short v;
        if (l16 == 0)      v = 0x3F80;                                       // 1.0 bf16
        else if (l16 == 1) v = f32_to_bf16(y[b * T_DIM + st * 16 + kh8 + j]);
        else               v = 0;
        ayfrag[(size_t)idx * 8 + j] = v;
    }
}

// ---- 16-row step (diagonal / parity peel). Data lives in the UPPER K half so
// the block-scan constant uhi (k >= m+16) doubles as the 16-row scan matrix. --
// lamneg2 = -lambda * log2(e): exp(-lam*S) computed as exp2(lamneg2*S).
template <bool DIAG>
__device__ __forceinline__ void tile16_step(const v16bf& as0, const v16bf& as1,
                                            const v16bf& bt0, const v16bf& bt1,
                                            const v16bf& uhi,
                                            const unsigned short* __restrict__ ayp,
                                            float lamneg2, int l16, int hi,
                                            float& carry, v8f& d2) {
    v8f c = {};
    c = __builtin_amdgcn_wmma_f32_16x16x32_bf16(false, as0, false, bt0, (short)0, c, false, false);
    c = __builtin_amdgcn_wmma_f32_16x16x32_bf16(false, as1, false, bt1, (short)0, c, false, false);
    if (DIAG) {
#pragma unroll
        for (int i = 0; i < 8; ++i)                    // keep strictly past: s < t
            c[i] = ((i + 8 * hi) < l16) ? c[i] : 0.0f;
    }
    v16bf bscan;
#pragma unroll
    for (int j = 0; j < 8; ++j) { bscan[j] = (__bf16)0.0f; bscan[j + 8] = (__bf16)c[j]; }
    v8f d = {};
    d = __builtin_amdgcn_wmma_f32_16x16x32_bf16(false, uhi, false, bscan, (short)0, d, false, false);

    const v8bf ayv = *(const v8bf*)ayp;
    const float cneg = lamneg2 * carry;
    v16bf wp, ay;
#pragma unroll
    for (int i = 0; i < 8; ++i) {
        wp[i] = (__bf16)0.0f;
        wp[i + 8] = (__bf16)(c[i] * __builtin_amdgcn_exp2f(__builtin_fmaf(lamneg2, d[i], cneg)));
        ay[i] = (__bf16)0.0f;
        ay[i + 8] = ayv[i];
    }
    d2 = __builtin_amdgcn_wmma_f32_16x16x32_bf16(false, ay, false, wp, (short)0, d2, false, false);
    carry += __shfl(d[0], l16, 32);      // tile total (row 0) from hi=0 partner
}

// ---- 32-row block step: 4 sim WMMA + 2 full-K scan WMMA + 1 full-K reduce ----
__device__ __forceinline__ void block32_step(const v16bf& al0, const v16bf& al1,   // low 16 rows
                                             const v16bf& ah0, const v16bf& ah1,   // high 16 rows
                                             const v16bf& bt0, const v16bf& bt1,
                                             const v16bf& ulo, const v16bf& uhi,
                                             const unsigned short* __restrict__ ayplo,
                                             const unsigned short* __restrict__ ayphi,
                                             float lamneg2, int l16,
                                             float& carry, v8f& d2) {
    v8f clo = {}, chi = {};
    clo = __builtin_amdgcn_wmma_f32_16x16x32_bf16(false, al0, false, bt0, (short)0, clo, false, false);
    clo = __builtin_amdgcn_wmma_f32_16x16x32_bf16(false, al1, false, bt1, (short)0, clo, false, false);
    chi = __builtin_amdgcn_wmma_f32_16x16x32_bf16(false, ah0, false, bt0, (short)0, chi, false, false);
    chi = __builtin_amdgcn_wmma_f32_16x16x32_bf16(false, ah1, false, bt1, (short)0, chi, false, false);

    v16bf bpack;                          // B operand: block rows 0..31 along K
#pragma unroll
    for (int j = 0; j < 8; ++j) { bpack[j] = (__bf16)clo[j]; bpack[j + 8] = (__bf16)chi[j]; }
    v8f dlo = {}, dhi = {};               // suffix sums over the full 32-row block
    dlo = __builtin_amdgcn_wmma_f32_16x16x32_bf16(false, ulo, false, bpack, (short)0, dlo, false, false);
    dhi = __builtin_amdgcn_wmma_f32_16x16x32_bf16(false, uhi, false, bpack, (short)0, dhi, false, false);

    const v8bf aylo = *(const v8bf*)ayplo;
    const v8bf ayhi = *(const v8bf*)ayphi;
    const float cneg = lamneg2 * carry;
    v16bf wp, ay;
#pragma unroll
    for (int i = 0; i < 8; ++i) {
        wp[i]     = (__bf16)(clo[i] * __builtin_amdgcn_exp2f(__builtin_fmaf(lamneg2, dlo[i], cneg)));
        wp[i + 8] = (__bf16)(chi[i] * __builtin_amdgcn_exp2f(__builtin_fmaf(lamneg2, dhi[i], cneg)));
        ay[i] = aylo[i]; ay[i + 8] = ayhi[i];
    }
    d2 = __builtin_amdgcn_wmma_f32_16x16x32_bf16(false, ay, false, wp, (short)0, d2, false, false);
    carry += __shfl(dlo[0], l16, 32);     // 32-row block total from hi=0 partner
}

// ---- Kernel 2: one wave32 per (b, 16-col t-tile); diag + peel + 32-row blocks --
__global__ __launch_bounds__(256) void ektc_main(const unsigned short* __restrict__ xh,
                                                 const unsigned short* __restrict__ ayfrag,
                                                 const float* __restrict__ lam,
                                                 const float* __restrict__ guess,
                                                 const float* __restrict__ slip,
                                                 float* __restrict__ out) {
    const int lane = threadIdx.x & 31;
    const int wid  = threadIdx.x >> 5;
    const int task = blockIdx.x * 8 + wid;       // 0 .. B*NTILE-1
    const int b    = task >> 7;                  // / NTILE
    const int tt   = task & (NTILE - 1);
    const int l16  = lane & 15;
    const int hi   = lane >> 4;
    const int kh8  = hi * 8;
    const int t_base = tt * 16;
    const unsigned short* xb  = xh + (size_t)b * T_DIM * K_DIM;
    const unsigned short* ayb = ayfrag + ((size_t)b * NTILE * 32 + lane) * 8;

    // B operand (t-columns), loop-invariant
    const v16bf bt0 = load_frag(xb + (size_t)(t_base + l16) * K_DIM,      kh8); // k 0..31
    const v16bf bt1 = load_frag(xb + (size_t)(t_base + l16) * K_DIM + 32, kh8); // k 32..63

    // scan constants over a 32-row block: ulo[m,k] = (k>=m), uhi[m,k] = (k>=m+16)
    v16bf ulo, uhi;
#pragma unroll
    for (int j = 0; j < 8; ++j) {
        const __bf16 tri = ((kh8 + j) >= l16) ? (__bf16)1.0f : (__bf16)0.0f;
        ulo[j] = tri;          ulo[j + 8] = (__bf16)1.0f;   // k>=16 always >= m(<16)
        uhi[j] = (__bf16)0.0f; uhi[j + 8] = tri;            // (k-16) >= m
    }

    // fold log2(e) into lambda: exp(-lam*S) == exp2(lamneg2*S)
    const float lamneg2 = -LOG2E * lam[b * T_DIM + t_base + l16];
    float carry = 0.f;
    v8f d2 = {};                                           // rows: 0 -> sum w, 1 -> sum y*w

    // diagonal tile: s-rows == t-rows -> A fragments are the B fragments
    tile16_step<true>(bt0, bt1, bt0, bt1, uhi, ayb + (size_t)tt * 256,
                      lamneg2, l16, hi, carry, d2);

    int p = tt - 1;                       // highest remaining 16-tile
    if (tt & 1) {                         // parity peel so the rest pairs evenly
        const unsigned short* srow = xb + (size_t)(p * 16 + l16) * K_DIM;
        const v16bf as0 = load_frag(srow,      kh8);
        const v16bf as1 = load_frag(srow + 32, kh8);
        tile16_step<false>(as0, as1, bt0, bt1, uhi, ayb + (size_t)p * 256,
                           lamneg2, l16, hi, carry, d2);
        --p;
    }
    for (; p > 0; p -= 2) {               // 32-row blocks {p-1, p}, s descending
        const unsigned short* rlo = xb + (size_t)((p - 1) * 16 + l16) * K_DIM;
        const unsigned short* rhi = xb + (size_t)(p * 16 + l16) * K_DIM;
        const v16bf al0 = load_frag(rlo,      kh8);
        const v16bf al1 = load_frag(rlo + 32, kh8);
        const v16bf ah0 = load_frag(rhi,      kh8);
        const v16bf ah1 = load_frag(rhi + 32, kh8);
        if (p > 2) __builtin_prefetch(rlo - 32 * K_DIM, 0, 1);   // next block
        block32_step(al0, al1, ah0, ah1, bt0, bt1, ulo, uhi,
                     ayb + (size_t)(p - 1) * 256, ayb + (size_t)p * 256,
                     lamneg2, l16, carry, d2);
    }

    if (hi == 0) {                        // d2[0]=sum w, d2[1]=sum y*w for t=t_base+l16
        const int idx = b * T_DIM + t_base + l16;
        const float h = d2[1] / (d2[0] + 1e-6f);
        const float g = guess[idx];
        const float s = slip[idx];
        float yh = h * (1.0f - s) + (1.0f - h) * g;
        out[idx] = fminf(fmaxf(yh, 0.01f), 0.99f);
    }
}

extern "C" void kernel_launch(void* const* d_in, const int* in_sizes, int n_in,
                              void* d_out, int out_size, void* d_ws, size_t ws_size,
                              hipStream_t stream) {
    const float* y     = (const float*)d_in[0];
    const int*   pseq  = (const int*)  d_in[1];
    const float* A     = (const float*)d_in[2];
    const float* kcw   = (const float*)d_in[3];
    const float* lamw  = (const float*)d_in[4];
    const float* lamb  = (const float*)d_in[5];
    const float* gw    = (const float*)d_in[6];
    const float* gb    = (const float*)d_in[7];
    const float* sw    = (const float*)d_in[8];
    const float* sb    = (const float*)d_in[9];
    float* out = (float*)d_out;

    // workspace layout (~5.6 MB):
    char* ws = (char*)d_ws;
    unsigned short* xh = (unsigned short*)ws;                     // B*T*64 bf16 = 4 MiB
    float* lam   = (float*)(ws + (size_t)B_DIM * T_DIM * K_DIM * sizeof(unsigned short));
    float* guess = lam   + B_DIM * T_DIM;
    float* slip  = guess + B_DIM * T_DIM;
    float* vlam  = slip  + B_DIM * T_DIM;
    float* vg    = vlam  + K_DIM;
    float* vs    = vg    + K_DIM;
    unsigned short* ayfrag = (unsigned short*)(vs + K_DIM);       // B*NTILE*32*8 bf16 = 1 MiB

    ektc_proj<<<1, 64, 0, stream>>>(kcw, lamw, gw, sw, vlam, vg, vs);
    ektc_gather<<<(B_DIM * T_DIM + 255) / 256, 256, 0, stream>>>(
        pseq, A, vlam, vg, vs, lamb, gb, sb, xh, lam, guess, slip);
    ektc_yfrag<<<(B_DIM * NTILE * 32 + 255) / 256, 256, 0, stream>>>(y, ayfrag);
    // B*NTILE = 2048 wave-tasks, 8 waves (256 threads) per block -> 256 blocks
    ektc_main<<<(B_DIM * NTILE) / 8, 256, 0, stream>>>(xh, ayfrag, lam, guess, slip, out);
}
